// AttentionLayer_7430293422092
// MI455X (gfx1250) — compile-verified, matrix operand order checked by reading, not run on hardware
//
#include <hip/hip_runtime.h>
#include <hip/hip_bf16.h>

// ---------------------------------------------------------------------------
// Fused attention layer for MI455X (gfx1250, wave32, WMMA + async-to-LDS).
//   q = Wq x + bq ; k = Wk x + bk ; v = Wv x + bv   (1x1 convs -> GEMMs)
//   out = v @ softmax(q^T k)^T + x                  (flash-style, fused)
// GEMMs run on v_wmma_f32_16x16x32_f16 (d=32 == one WMMA K step).
// K/V j-tiles are staged into LDS with GLOBAL_LOAD_ASYNC_TO_LDS_B128
// (ASYNCcnt-tracked, double-buffered) so the 4 waves of a block share them.
// ---------------------------------------------------------------------------

typedef __attribute__((ext_vector_type(16))) _Float16 v16h;
typedef __attribute__((ext_vector_type(8)))  _Float16 v8h;
typedef __attribute__((ext_vector_type(8)))  float    v8f;
typedef __attribute__((ext_vector_type(4)))  float    v4f;

#define BATCH 4
#define CH    256
#define DQK   32      // = CH/8
#define NPIX  4096    // 64*64

// Load a 16-bit WMMA A/B fragment (16x32 / 32x16). Per the CDNA5 ISA layout,
// lane L holds row/col (L&15); halves 0..7 live at K-offset koff, halves
// 8..15 at koff+16, where koff = (L<16 ? 0 : 8). Two 16B loads per lane.
__device__ __forceinline__ v16h load_frag(const _Float16* base, int koff) {
  union { v16h v; v8h h[2]; } u;
  u.h[0] = *(const v8h*)(base + koff);
  u.h[1] = *(const v8h*)(base + koff + 16);
  return u.v;
}

__device__ __forceinline__ v8f wmma_f16(v16h a, v16h b, v8f c) {
  // (neg_a, A, neg_b, B, c_mod, C, reuse_a, reuse_b)
  return __builtin_amdgcn_wmma_f32_16x16x32_f16(false, a, false, b, (short)0, c,
                                                false, false);
}

// Async copy of one 16B chunk global -> LDS (tracked by ASYNCcnt, not VGPRs).
// lds_off is a byte offset from the workgroup LDS base (low 32 bits of a
// generic pointer to a __shared__ object).
__device__ __forceinline__ void async_copy16(unsigned lds_off, const _Float16* g) {
  asm volatile("global_load_async_to_lds_b128 %0, %1, off"
               :: "v"(lds_off), "v"((unsigned long long)(uintptr_t)g)
               : "memory");
}

__device__ __forceinline__ unsigned lds_off_of(const void* p) {
  return (unsigned)(uintptr_t)p;   // addrspace(3) offset lives in the low 32b
}

// ---------------------------------------------------------------------------
// Kernel 1: pack weights (wq||wk||wv -> Wh[320][256] f16) + bias vector.
// ---------------------------------------------------------------------------
__global__ void cvt_w_kernel(const float* __restrict__ wq, const float* __restrict__ bq,
                             const float* __restrict__ wk, const float* __restrict__ bk,
                             const float* __restrict__ wv, const float* __restrict__ bv,
                             _Float16* __restrict__ Wh, float* __restrict__ Bias) {
  int idx = blockIdx.x * 256 + threadIdx.x;          // 320 blocks * 256
  int o = idx >> 8, c = idx & 255;
  float w = (o < 32) ? wq[o * 256 + c]
          : (o < 64) ? wk[(o - 32) * 256 + c]
                     : wv[(o - 64) * 256 + c];
  Wh[idx] = (_Float16)w;
  if (idx < 320)
    Bias[idx] = (idx < 32) ? bq[idx] : (idx < 64) ? bk[idx - 32] : bv[idx - 64];
}

// ---------------------------------------------------------------------------
// Kernel 2: transpose+convert x[b][c][n] f32 -> XTh[b][n][c] f16 (LDS tiles).
// ---------------------------------------------------------------------------
__global__ void cvt_x_kernel(const float* __restrict__ x, _Float16* __restrict__ XT) {
  __shared__ float tile[32][33];
  const int b = blockIdx.z;
  const int n0 = blockIdx.x * 32, c0 = blockIdx.y * 32;
  const float* xb = x + ((size_t)b * CH + c0) * NPIX + n0;
  #pragma unroll
  for (int i = 0; i < 32; i += 8)
    tile[threadIdx.y + i][threadIdx.x] = xb[(size_t)(threadIdx.y + i) * NPIX + threadIdx.x];
  __syncthreads();
  _Float16* xt = XT + ((size_t)b * NPIX + n0) * CH + c0;
  #pragma unroll
  for (int i = 0; i < 32; i += 8)
    xt[(size_t)(threadIdx.y + i) * CH + threadIdx.x] =
        (_Float16)tile[threadIdx.x][threadIdx.y + i];
}

// ---------------------------------------------------------------------------
// Kernel 3: projection GEMM  [320 x 256] @ [256 x 4096]  per batch, via WMMA.
//   Qh[b][n][32], Kh[b][n][32]  (row-major per pixel -> fragment friendly)
//   Vh[b][c][n]                 (j contiguous -> PV B-fragment friendly)
// One wave per 16-wide i-tile; X fragments hoisted across all 20 o-tiles.
// ---------------------------------------------------------------------------
__global__ __launch_bounds__(128) void proj_kernel(
    const _Float16* __restrict__ Wh, const float* __restrict__ Bias,
    const _Float16* __restrict__ XTh,
    _Float16* __restrict__ Qh, _Float16* __restrict__ Kh, _Float16* __restrict__ Vh) {
  const int lane  = threadIdx.x & 31;
  const int wave  = threadIdx.x >> 5;
  const int b     = blockIdx.y;
  const int i0    = (blockIdx.x * 4 + wave) * 16;
  const int half_ = lane >> 4;
  const int l15   = lane & 15;
  const int koff  = half_ * 8;

  // Hoist the 8 K-chunks of X for this i-tile (B operands).
  v16h Bx[8];
  const _Float16* xtrow = XTh + ((size_t)b * NPIX + i0 + l15) * CH;
  #pragma unroll
  for (int k = 0; k < 8; ++k) Bx[k] = load_frag(xtrow + k * 32, koff);

  #pragma unroll 1
  for (int ot = 0; ot < 20; ++ot) {           // 20 o-tiles: 2 Q, 2 K, 16 V
    v8f acc;
    #pragma unroll
    for (int r = 0; r < 8; ++r) acc[r] = 0.0f;
    const _Float16* wrow = Wh + (size_t)(ot * 16 + l15) * CH;
    #pragma unroll
    for (int k = 0; k < 8; ++k) {
      v16h Aw = load_frag(wrow + k * 32, koff);
      acc = wmma_f16(Aw, Bx[k], acc);
    }
    const int obase = ot * 16 + 8 * half_;    // D rows: M = r + 8*half_
    #pragma unroll
    for (int r = 0; r < 8; ++r) acc[r] += Bias[obase + r];

    const int i = i0 + l15;                   // D cols: N = l15
    if (ot < 4) {                             // Q / K : rows r contiguous in o
      _Float16* dst = (ot < 2) ? Qh : Kh;
      const int oo = (ot & 1) * 16 + 8 * half_;
      v8h pk;
      #pragma unroll
      for (int r = 0; r < 8; ++r) pk[r] = (_Float16)acc[r];
      *(v8h*)(dst + ((size_t)b * NPIX + i) * DQK + oo) = pk;
    } else {                                  // V : [c][n], scatter 8 halves
      const int c0 = (ot - 4) * 16 + 8 * half_;
      #pragma unroll
      for (int r = 0; r < 8; ++r)
        Vh[((size_t)b * CH + c0 + r) * NPIX + i] = (_Float16)acc[r];
    }
  }
}

// ---------------------------------------------------------------------------
// Kernel 4: fused flash attention + residual.
// Block = 4 waves, each wave owns 16 query rows x all 256 channels
// (16 v8f accumulators). The block cooperatively double-buffers each 32-wide
// j-block of K (2KB) and V (16KB) into LDS with async-to-LDS copies (9 per
// thread per block-iteration, waited with s_wait_asynccnt), so K/V global
// traffic is shared 4 ways and overlapped with the 18 WMMAs per iteration.
// ---------------------------------------------------------------------------
__global__ __launch_bounds__(128) void attn_kernel(
    const _Float16* __restrict__ Qh, const _Float16* __restrict__ Kh,
    const _Float16* __restrict__ Vh, const float* __restrict__ x,
    float* __restrict__ out) {
  __shared__ __align__(16) _Float16 sV[2][CH * 32];   // 2 x 16KB: V[c][j] tile
  __shared__ __align__(16) _Float16 sK[2][32 * 32];   // 2 x  2KB: K[j][d] tile
  __shared__ __align__(16) _Float16 sP[4][16 * 32];   // 1KB per wave: P relayout

  const int tid   = threadIdx.x;
  const int lane  = tid & 31;
  const int wave  = tid >> 5;
  const int b     = blockIdx.y;
  const int i0    = (blockIdx.x * 4 + wave) * 16;
  const int half_ = lane >> 4;
  const int l15   = lane & 15;
  const int koff  = half_ * 8;
  _Float16* pbuf  = sP[wave];

  // Issue the 9 async 16B chunks this thread contributes for j-block jb.
  const _Float16* Kb_g = Kh + (size_t)b * NPIX * DQK;
  const _Float16* Vb_g = Vh + (size_t)b * CH * NPIX;
  auto issue_tile = [&](int jb, int p) {
    {  // K tile: 32 rows x 64B = 128 chunks (1 per thread)
      const int j = tid >> 2, part = tid & 3;
      async_copy16(lds_off_of(&sK[p][0]) + tid * 16,
                   Kb_g + (size_t)(jb + j) * DQK + part * 8);
    }
    #pragma unroll
    for (int q = 0; q < 8; ++q) {  // V tile: 256 rows x 64B = 1024 chunks
      const int chunk = q * 128 + tid;
      const int c = chunk >> 2, part = chunk & 3;
      async_copy16(lds_off_of(&sV[p][0]) + chunk * 16,
                   Vb_g + (size_t)c * NPIX + jb + part * 8);
    }
  };

  // Q fragment for this wave's 16 rows (reused for all 128 j-blocks).
  const v16h Qa = load_frag(Qh + ((size_t)b * NPIX + i0 + l15) * DQK, koff);

  v8f acc[16];
  #pragma unroll
  for (int t = 0; t < 16; ++t)
    #pragma unroll
    for (int r = 0; r < 8; ++r) acc[t][r] = 0.0f;

  float mrow[8], lrow[8];
  #pragma unroll
  for (int r = 0; r < 8; ++r) { mrow[r] = -3.0e38f; lrow[r] = 0.0f; }

  issue_tile(0, 0);   // prologue fill of buffer 0

  #pragma unroll 1
  for (int it = 0; it < NPIX / 32; ++it) {
    const int p  = it & 1;
    const int jb = it * 32;

    // Prefetch next j-block into the other buffer, then wait for this one.
    // Async loads complete in order, so <=9 outstanding => buffer p is ready.
    if (jb + 32 < NPIX) {
      issue_tile(jb + 32, p ^ 1);
      asm volatile("s_wait_asynccnt 0x9" ::: "memory");
    } else {
      asm volatile("s_wait_asynccnt 0x0" ::: "memory");
    }
    __syncthreads();

    // ---- scores: S = Q K^T for 16 rows x 32 cols (two 16x16 D tiles) ----
    const _Float16* kb = &sK[p][0];
    v16h Kb0 = load_frag(kb + l15 * 32, koff);
    v16h Kb1 = load_frag(kb + (16 + l15) * 32, koff);
    v8f z;
    #pragma unroll
    for (int r = 0; r < 8; ++r) z[r] = 0.0f;
    v8f s0 = wmma_f16(Qa, Kb0, z);
    v8f s1 = wmma_f16(Qa, Kb1, z);

    // ---- online softmax (per-row state replicated across 16-lane group) ----
    float p0[8], p1[8], arow[8];
    #pragma unroll
    for (int r = 0; r < 8; ++r) {
      float mx = fmaxf(s0[r], s1[r]);
      #pragma unroll
      for (int m = 1; m < 16; m <<= 1) mx = fmaxf(mx, __shfl_xor(mx, m, 16));
      const float mnew = fmaxf(mrow[r], mx);
      arow[r] = __expf(mrow[r] - mnew);
      mrow[r] = mnew;
      p0[r] = __expf(s0[r] - mnew);
      p1[r] = __expf(s1[r] - mnew);
      float sum = p0[r] + p1[r];
      #pragma unroll
      for (int m = 1; m < 16; m <<= 1) sum += __shfl_xor(sum, m, 16);
      lrow[r] = lrow[r] * arow[r] + sum;
    }
    #pragma unroll
    for (int t = 0; t < 16; ++t)
      #pragma unroll
      for (int r = 0; r < 8; ++r) acc[t][r] *= arow[r];

    // ---- D-layout P -> A-layout fragment via this wave's LDS slice ----
    #pragma unroll
    for (int r = 0; r < 8; ++r) {
      const int row = r + 8 * half_;
      pbuf[row * 32 + l15]      = (_Float16)p0[r];
      pbuf[row * 32 + l15 + 16] = (_Float16)p1[r];
    }
    v16h Pa = load_frag(pbuf + l15 * 32, koff);   // same-wave LDS: HW in-order

    // ---- O += P V^T over all 16 channel tiles (B fragments from LDS) ----
    const _Float16* vb = &sV[p][0];
    #pragma unroll
    for (int t = 0; t < 16; ++t) {
      v16h Vb = load_frag(vb + (t * 16 + l15) * 32, koff);
      acc[t] = wmma_f16(Pa, Vb, acc[t]);
    }
    __syncthreads();   // all waves done with buffer p before it is refilled
  }

  // ---- epilogue: out = acc/l + x, vectorized 32B stores ----
  float rl[8];
  #pragma unroll
  for (int r = 0; r < 8; ++r) rl[r] = 1.0f / lrow[r];
  const size_t obase = (size_t)b * CH * NPIX;
  const int irow = i0 + 8 * half_;
  #pragma unroll
  for (int t = 0; t < 16; ++t) {
    const int c = t * 16 + l15;
    const float* xp = x + obase + (size_t)c * NPIX + irow;
    float* op = out + obase + (size_t)c * NPIX + irow;
    v4f xa = *(const v4f*)xp;
    v4f xb2 = *(const v4f*)(xp + 4);
    v4f oa, ob;
    #pragma unroll
    for (int r = 0; r < 4; ++r) {
      oa[r] = acc[t][r] * rl[r] + xa[r];
      ob[r] = acc[t][r + 4] * rl[r + 4] + xb2[r];
    }
    *(v4f*)op = oa;
    *(v4f*)(op + 4) = ob;
  }
}

// ---------------------------------------------------------------------------
// Host-side launcher. Workspace layout (~18.5 MiB):
//   [0)        Wh   320*256 f16
//   [256K)     Bias 320 f32
//   [512K)     XTh  4*4096*256 f16   (8 MiB)
//   [+8M)      Qh   4*4096*32  f16   (1 MiB)
//   [+1M)      Kh   4*4096*32  f16   (1 MiB)
//   [+1M)      Vh   4*256*4096 f16   (8 MiB)
// ---------------------------------------------------------------------------
extern "C" void kernel_launch(void* const* d_in, const int* in_sizes, int n_in,
                              void* d_out, int out_size, void* d_ws, size_t ws_size,
                              hipStream_t stream) {
  const float* x  = (const float*)d_in[0];
  const float* wq = (const float*)d_in[1];
  const float* bq = (const float*)d_in[2];
  const float* wk = (const float*)d_in[3];
  const float* bk = (const float*)d_in[4];
  const float* wv = (const float*)d_in[5];
  const float* bv = (const float*)d_in[6];
  float* out = (float*)d_out;

  char* ws = (char*)d_ws;
  _Float16* Wh   = (_Float16*)(ws);
  float*    Bias = (float*)(ws + 256 * 1024);
  _Float16* XTh  = (_Float16*)(ws + 512 * 1024);
  _Float16* Qh   = (_Float16*)(ws + 512 * 1024 + 8ull * 1024 * 1024);
  _Float16* Kh   = (_Float16*)(ws + 512 * 1024 + 9ull * 1024 * 1024);
  _Float16* Vh   = (_Float16*)(ws + 512 * 1024 + 10ull * 1024 * 1024);

  cvt_w_kernel<<<320, 256, 0, stream>>>(wq, bq, wk, bk, wv, bv, Wh, Bias);
  cvt_x_kernel<<<dim3(NPIX / 32, CH / 32, BATCH), dim3(32, 8), 0, stream>>>(x, XTh);
  proj_kernel<<<dim3(NPIX / 64, BATCH), 128, 0, stream>>>(Wh, Bias, XTh, Qh, Kh, Vh);
  attn_kernel<<<dim3(NPIX / 64, BATCH), 128, 0, stream>>>(Qh, Kh, Vh, x, out);
}
